// EmbededNonLocalLayer_10788957847970
// MI455X (gfx1250) — matrix operand order repaired and checked by reading, not compiled, not run on hardware
//
#include <hip/hip_runtime.h>
#include <hip/hip_bf16.h>

// ---------------- problem constants ----------------
#define BATCH 4
#define C_IN  512
#define C_I   256
#define C_O   512
#define HW    63
#define NSP   (HW * HW)      // 3969
#define NP    4032           // padded N (63*64)
#define KKV   81
#define KP    96             // padded 81
#define ATT_SCALE 0.0625f    // CI^-0.5
#define BN_EPS    1e-5f
#define NT_NP 252            // NP/16
#define CH_NP 126            // NP/32

typedef __attribute__((ext_vector_type(16))) __bf16 v16bf;
typedef __attribute__((ext_vector_type(8)))  float  v8f;

// ---------------- WMMA + swizzled-operand helpers ----------------
__device__ __forceinline__ v8f wmma_bf16(v16bf a, v16bf b, v8f c) {
  return __builtin_amdgcn_wmma_f32_16x16x32_bf16(false, a, false, b, (short)0, c,
                                                 false, false);
}

// Operand tiles are stored pre-swizzled: 512 bf16 per 16x32 (A) or 32x16 (B) tile,
// lane L's 16 contiguous elements at tile + L*16  ->  one 32B vector load.
__device__ __forceinline__ v16bf load_sw(const __bf16* tile, int lane) {
  return *(const v16bf*)(tile + (size_t)lane * 16);
}

// A-register layout: lane = m + 16*(K>>4); elem idx = K&15
__device__ __forceinline__ int a_sw_idx(int m, int K) {
  return (m + 16 * ((K >> 4) & 1)) * 16 + (K & 15);
}
// B-register layout: lane = n + 16*(K>>4); elem idx = K&15
__device__ __forceinline__ int b_sw_idx(int K, int n) {
  return (n + 16 * ((K >> 4) & 1)) * 16 + (K & 15);
}

// reductions within each 16-lane half of a wave32 (C/D rows live in a half)
__device__ __forceinline__ float hmax16(float v) {
#pragma unroll
  for (int m = 1; m <= 8; m <<= 1) v = fmaxf(v, __shfl_xor(v, m, 32));
  return v;
}
__device__ __forceinline__ float hsum16(float v) {
#pragma unroll
  for (int m = 1; m <= 8; m <<= 1) v += __shfl_xor(v, m, 32);
  return v;
}

// CDNA5 async global->LDS copy (ASYNCcnt tracked): 16 bytes per lane, no VGPR bounce.
__device__ __forceinline__ void async_b128_to_lds(void* lds_ptr, const void* gptr) {
  unsigned loff = (unsigned)(size_t)lds_ptr;  // LDS aperture: low 32 bits = LDS byte addr
  asm volatile("global_load_async_to_lds_b128 %0, %1, off"
               :: "v"(loff), "v"(gptr) : "memory");
}
__device__ __forceinline__ void wait_asynccnt0() {
  asm volatile("s_wait_asynccnt 0x0" ::: "memory");
}

// ---------------- workspace layout (bytes) ----------------
constexpr size_t SZ_W_BIG  = (size_t)C_I * C_IN * 2;
constexpr size_t SZ_WV2    = (size_t)C_I * C_I * 2;
constexpr size_t SZ_WW     = (size_t)C_O * C_I * 2;
constexpr size_t OFF_WKA   = 0;
constexpr size_t OFF_WVA   = OFF_WKA + SZ_W_BIG;
constexpr size_t OFF_WV2A  = OFF_WVA + SZ_W_BIG;
constexpr size_t OFF_WWA   = OFF_WV2A + SZ_WV2;
constexpr size_t OFF_INT   = OFF_WWA + SZ_WW;
constexpr size_t SZ_VALF   = (size_t)BATCH * C_I * NP * 4;
constexpr size_t SZ_NPCI   = (size_t)BATCH * NT_NP * 8 * 512 * 2;   // = B*NP*CI*2
constexpr size_t SZ_POOL   = (size_t)BATCH * C_I * KP * 2;
constexpr size_t SZ_SIMV   = (size_t)BATCH * NP * KP * 2;
constexpr size_t OFF_VALF  = OFF_INT;
constexpr size_t OFF_VA    = OFF_VALF + SZ_VALF;
constexpr size_t OFF_QKA   = OFF_VA + SZ_NPCI;
constexpr size_t OFF_QKB   = OFF_QKA + SZ_NPCI;
constexpr size_t OFF_POOL  = OFF_QKB + SZ_NPCI;
constexpr size_t OFF_V2B   = OFF_POOL + SZ_POOL;
constexpr size_t OFF_V2A   = OFF_V2B + SZ_POOL;
constexpr size_t OFF_SIMV  = OFF_V2A + SZ_POOL;
constexpr size_t OFF_SNT   = OFF_SIMV + SZ_SIMV;
constexpr size_t OFF_END   = OFF_SNT + SZ_SIMV;

// ---------------- kernel 0: weight conversion + A-swizzle ----------------
__global__ void k0_prep(const float* __restrict__ Wk, const float* __restrict__ Wv,
                        const float* __restrict__ Wv2, const float* __restrict__ Ww,
                        __bf16* __restrict__ WkA, __bf16* __restrict__ WvA,
                        __bf16* __restrict__ Wv2A, __bf16* __restrict__ WwA) {
  int i = blockIdx.x * blockDim.x + threadIdx.x;
  if (i < C_I * C_IN) {
    int m = i / C_IN, K = i % C_IN;
    size_t off = (size_t)((m >> 4) * (C_IN / 32) + (K >> 5)) * 512 + a_sw_idx(m & 15, K & 31);
    WkA[off] = (__bf16)Wk[i];
    WvA[off] = (__bf16)Wv[i];
  }
  if (i < C_I * C_I) {
    int m = i / C_I, K = i % C_I;
    size_t off = (size_t)((m >> 4) * 8 + (K >> 5)) * 512 + a_sw_idx(m & 15, K & 31);
    Wv2A[off] = (__bf16)Wv2[i];
  }
  if (i < C_O * C_I) {
    int m = i / C_I, K = i % C_I;
    size_t off = (size_t)((m >> 4) * 8 + (K >> 5)) * 512 + a_sw_idx(m & 15, K & 31);
    WwA[off] = (__bf16)Ww[i];
  }
}

__global__ void k0_zero(uint4* __restrict__ p, size_t n16) {
  size_t i = (size_t)blockIdx.x * blockDim.x + threadIdx.x;
  size_t stride = (size_t)gridDim.x * blockDim.x;
  uint4 z; z.x = 0; z.y = 0; z.z = 0; z.w = 0;
  for (; i < n16; i += stride) p[i] = z;
}

// ---------------- kernel 1: value & qk 1x1 convs (dual GEMM) ----------------
__global__ __launch_bounds__(256) void k1_qkv(
    const float* __restrict__ x, const __bf16* __restrict__ WvA,
    const __bf16* __restrict__ WkA, const float* __restrict__ bv,
    const float* __restrict__ bk, const float* __restrict__ gamma,
    const float* __restrict__ beta, float* __restrict__ value_f,
    __bf16* __restrict__ vAsw, __bf16* __restrict__ qkAsw, __bf16* __restrict__ qkBsw) {
  __shared__ __bf16 lx[C_IN * 64];  // 16 kc x 4 nt swizzled B tiles
  const int b = blockIdx.y, n0 = blockIdx.x * 64, tid = threadIdx.x;
  // stage x tile as B-swizzled bf16 (pack 2 consecutive-K bf16 per b32 store)
#pragma unroll 1
  for (int e = tid; e < (C_IN / 2) * 64; e += 256) {
    int j = e & 63, c = (e >> 6) * 2;
    int n = n0 + j;
    float x0 = 0.f, x1 = 0.f;
    if (n < NSP) {
      x0 = x[((size_t)b * C_IN + c) * NSP + n];
      x1 = x[((size_t)b * C_IN + c + 1) * NSP + n];
    }
    __bf16 v0 = (__bf16)x0, v1 = (__bf16)x1;
    unsigned int pk = ((unsigned int)__builtin_bit_cast(unsigned short, v1) << 16) |
                      (unsigned int)__builtin_bit_cast(unsigned short, v0);
    int off = ((c >> 5) * 4 + (j >> 4)) * 512 + ((j & 15) + 16 * ((c >> 4) & 1)) * 16 + (c & 15);
    *(unsigned int*)(lx + off) = pk;
  }
  __syncthreads();
  const int wave = tid >> 5, lane = tid & 31;
  const int mt0 = wave * 2;  // two 16-row output tiles per wave
  v8f cv[2][4] = {}, cq[2][4] = {};
#pragma unroll 1
  for (int kc = 0; kc < 16; ++kc) {
    v16bf aV[2], aK[2];
#pragma unroll
    for (int mt = 0; mt < 2; ++mt) {
      aV[mt] = load_sw(WvA + (size_t)((mt0 + mt) * 16 + kc) * 512, lane);
      aK[mt] = load_sw(WkA + (size_t)((mt0 + mt) * 16 + kc) * 512, lane);
    }
#pragma unroll
    for (int nt = 0; nt < 4; ++nt) {
      v16bf bB = load_sw(lx + (size_t)(kc * 4 + nt) * 512, lane);
#pragma unroll
      for (int mt = 0; mt < 2; ++mt) {
        cv[mt][nt] = wmma_bf16(aV[mt], bB, cv[mt][nt]);
        cq[mt][nt] = wmma_bf16(aK[mt], bB, cq[mt][nt]);
      }
    }
  }
  const float isq = rsqrtf(1.f + BN_EPS);
  const int ml = lane & 15, h = lane >> 4;
#pragma unroll
  for (int mt = 0; mt < 2; ++mt)
#pragma unroll
    for (int r = 0; r < 8; ++r) {
      int ch = wave * 32 + mt * 16 + r + 8 * h;
      float bvv = bv[ch], bkk = bk[ch];
      float bns = gamma[ch] * isq, bnb = beta[ch];
#pragma unroll
      for (int nt = 0; nt < 4; ++nt) {
        int n = n0 + nt * 16 + ml;
        if (n < NSP) {
          float vval = cv[mt][nt][r] + bvv;
          float qval = (cq[mt][nt][r] + bkk) * bns + bnb;
          value_f[((size_t)b * C_I + ch) * NP + n] = vval;
          size_t tbase = ((size_t)(b * NT_NP + (n >> 4)) * 8 + (ch >> 5)) * 512;
          int aoff = a_sw_idx(n & 15, ch & 31);
          vAsw[tbase + aoff]  = (__bf16)vval;
          qkAsw[tbase + aoff] = (__bf16)qval;
          qkBsw[tbase + b_sw_idx(ch & 31, n & 15)] = (__bf16)qval;
        }
      }
    }
}

// ---------------- kernel 2: exact 7x7 average pool -> B-swizzled bf16 ----------------
__global__ void k2_pool(const float* __restrict__ value_f, __bf16* __restrict__ pooledB) {
  int idx = blockIdx.x * blockDim.x + threadIdx.x;
  if (idx >= BATCH * C_I * KP) return;
  int k = idx % KP, c = (idx / KP) % C_I, b = idx / (KP * C_I);
  float acc = 0.f;
  if (k < KKV) {
    int ky = k / 9, kx = k % 9;
    const float* src = value_f + ((size_t)b * C_I + c) * NP;
#pragma unroll
    for (int i = 0; i < 7; ++i)
#pragma unroll
      for (int j = 0; j < 7; ++j)
        acc += src[(ky * 7 + i) * HW + kx * 7 + j];
    acc *= (1.f / 49.f);
  }
  size_t off = ((size_t)(b * 8 + (c >> 5)) * 6 + (k >> 4)) * 512 + b_sw_idx(c & 31, k & 15);
  pooledB[off] = (__bf16)acc;
}

// ---------------- kernel 3: v2 = Wv2 x pooled + bv2 (write A- and B-swizzled) ----------------
__global__ __launch_bounds__(256) void k3_v2(
    const __bf16* __restrict__ Wv2A, const __bf16* __restrict__ pooledB,
    const float* __restrict__ bv2, __bf16* __restrict__ v2B, __bf16* __restrict__ v2A) {
  int b = blockIdx.x, tid = threadIdx.x;
  int wave = tid >> 5, lane = tid & 31, ml = lane & 15, h = lane >> 4;
  int mt0 = wave * 2;
  v8f acc[2][6] = {};
#pragma unroll 1
  for (int kc = 0; kc < 8; ++kc) {
    v16bf aA[2];
#pragma unroll
    for (int mt = 0; mt < 2; ++mt)
      aA[mt] = load_sw(Wv2A + (size_t)((mt0 + mt) * 8 + kc) * 512, lane);
#pragma unroll
    for (int nt = 0; nt < 6; ++nt) {
      v16bf bB = load_sw(pooledB + ((size_t)(b * 8 + kc) * 6 + nt) * 512, lane);
#pragma unroll
      for (int mt = 0; mt < 2; ++mt) acc[mt][nt] = wmma_bf16(aA[mt], bB, acc[mt][nt]);
    }
  }
#pragma unroll
  for (int mt = 0; mt < 2; ++mt)
#pragma unroll
    for (int r = 0; r < 8; ++r) {
      int ch = wave * 32 + mt * 16 + r + 8 * h;
      float bb = bv2[ch];
#pragma unroll
      for (int nt = 0; nt < 6; ++nt) {
        int kcol = nt * 16 + ml;
        float val = acc[mt][nt][r] + bb;
        v2B[((size_t)(b * 8 + (ch >> 5)) * 6 + (kcol >> 4)) * 512 + b_sw_idx(ch & 31, kcol & 15)] =
            (__bf16)val;
        v2A[((size_t)(b * 16 + (ch >> 4)) * 3 + (kcol >> 5)) * 512 + a_sw_idx(ch & 15, kcol & 31)] =
            (__bf16)val;
      }
    }
}

// ---------------- kernel 4: simv = softmax_k( v^T v2 * scale ) -> B-swizzled ----------------
__global__ __launch_bounds__(128) void k4_simv(
    const __bf16* __restrict__ vAsw, const __bf16* __restrict__ v2B,
    __bf16* __restrict__ simvB) {
  int b = blockIdx.y, n0 = blockIdx.x * 64, tid = threadIdx.x;
  int wave = tid >> 5, lane = tid & 31, ml = lane & 15, h = lane >> 4;
  int m0 = n0 + wave * 16;
  v16bf aQ[8];
#pragma unroll
  for (int kc = 0; kc < 8; ++kc)
    aQ[kc] = load_sw(vAsw + ((size_t)(b * NT_NP + (m0 >> 4)) * 8 + kc) * 512, lane);
  v8f s[6] = {};
#pragma unroll
  for (int nt = 0; nt < 6; ++nt)
#pragma unroll
    for (int kc = 0; kc < 8; ++kc) {
      v16bf bB = load_sw(v2B + ((size_t)(b * 8 + kc) * 6 + nt) * 512, lane);
      s[nt] = wmma_bf16(aQ[kc], bB, s[nt]);
    }
#pragma unroll
  for (int r = 0; r < 8; ++r) {
    float mx = -1e30f;
#pragma unroll
    for (int nt = 0; nt < 6; ++nt) {
      float xv = s[nt][r] * ATT_SCALE;
      if (nt * 16 + ml >= KKV) xv = -1e30f;
      s[nt][r] = xv;
      mx = fmaxf(mx, xv);
    }
    mx = hmax16(mx);
    float sum = 0.f;
#pragma unroll
    for (int nt = 0; nt < 6; ++nt) {
      float p = __expf(s[nt][r] - mx);
      s[nt][r] = p;
      sum += p;
    }
    sum = hsum16(sum);
    float inv = 1.f / sum;
    int q = m0 + r + 8 * h;
#pragma unroll
    for (int nt = 0; nt < 6; ++nt) {
      int kcol = nt * 16 + ml;
      simvB[((size_t)(b * CH_NP + (q >> 5)) * 6 + (kcol >> 4)) * 512 +
            b_sw_idx(q & 31, kcol & 15)] = (__bf16)(s[nt][r] * inv);
    }
  }
}

// ---------------- kernel 5: fused flash attention (async double-buffered K/V) ----------------
__global__ __launch_bounds__(128) void k5_attn(
    const __bf16* __restrict__ qkAsw, const __bf16* __restrict__ qkBsw,
    const __bf16* __restrict__ simvB, __bf16* __restrict__ snTB) {
  __shared__ __bf16 ldsK[2][16 * 512];  // double-buffered K tiles (2 x 16 KB)
  __shared__ __bf16 ldsV[2][6 * 512];   // double-buffered V tiles (2 x 6 KB)
  __shared__ __bf16 ldsP[4][512];       // per-wave P in A-swizzled order
  int b = blockIdx.y, m0 = blockIdx.x * 64, tid = threadIdx.x;
  int wave = tid >> 5, lane = tid & 31, ml = lane & 15, h = lane >> 4;
  int mw = m0 + wave * 16;
  v16bf aQ[8];
#pragma unroll
  for (int kc = 0; kc < 8; ++kc)
    aQ[kc] = load_sw(qkAsw + ((size_t)(b * NT_NP + (mw >> 4)) * 8 + kc) * 512, lane);
  float run_m[8], run_l[8];
#pragma unroll
  for (int r = 0; r < 8; ++r) { run_m[r] = -1e30f; run_l[r] = 0.f; }
  v8f O[6] = {};
  __bf16* tp = &ldsP[wave][0];
  const __bf16* srcK0 = qkBsw + (size_t)b * NT_NP * 8 * 512;
  const __bf16* srcV0 = simvB + (size_t)b * CH_NP * 6 * 512;

  // prime buffer 0 with chunk 0 (async global->LDS, no VGPR bounce)
  {
    const __bf16* sK = srcK0;
    const __bf16* sV = srcV0;
#pragma unroll
    for (int g = 0; g < 8; ++g) {
      int idx = (tid + g * 128) * 8;
      async_b128_to_lds(&ldsK[0][idx], sK + idx);
    }
#pragma unroll
    for (int g = 0; g < 3; ++g) {
      int idx = (tid + g * 128) * 8;
      async_b128_to_lds(&ldsV[0][idx], sV + idx);
    }
  }
  wait_asynccnt0();
  __syncthreads();

#pragma unroll 1
  for (int ck = 0; ck < CH_NP; ++ck) {
    int nk0 = ck * 32;
    int cur = ck & 1;
    // prefetch next chunk into the other buffer while we compute on this one
    if (ck + 1 < CH_NP) {
      const __bf16* sK = srcK0 + (size_t)(ck + 1) * 16 * 512;
      const __bf16* sV = srcV0 + (size_t)(ck + 1) * 6 * 512;
#pragma unroll
      for (int g = 0; g < 8; ++g) {
        int idx = (tid + g * 128) * 8;
        async_b128_to_lds(&ldsK[cur ^ 1][idx], sK + idx);
      }
#pragma unroll
      for (int g = 0; g < 3; ++g) {
        int idx = (tid + g * 128) * 8;
        async_b128_to_lds(&ldsV[cur ^ 1][idx], sV + idx);
      }
    }
    // S = Q K^T over CI=256 (two 16-key sub-tiles)
    float sv[2][8];
#pragma unroll
    for (int sub = 0; sub < 2; ++sub) {
      v8f s = {};
#pragma unroll
      for (int kc = 0; kc < 8; ++kc) {
        v16bf bB = load_sw(&ldsK[cur][(size_t)(sub * 8 + kc) * 512], lane);
        s = wmma_bf16(aQ[kc], bB, s);
      }
      bool valid = (nk0 + sub * 16 + ml) < NSP;
#pragma unroll
      for (int r = 0; r < 8; ++r) sv[sub][r] = valid ? s[r] * ATT_SCALE : -1e30f;
    }
    // online softmax update
#pragma unroll
    for (int r = 0; r < 8; ++r) {
      float cm = hmax16(fmaxf(sv[0][r], sv[1][r]));
      float nm = fmaxf(run_m[r], cm);
      float al = __expf(run_m[r] - nm);
      run_m[r] = nm;
      run_l[r] *= al;
#pragma unroll
      for (int j = 0; j < 6; ++j) O[j][r] *= al;
      float p0 = __expf(sv[0][r] - nm);
      float p1 = __expf(sv[1][r] - nm);
      sv[0][r] = p0; sv[1][r] = p1;
      run_l[r] += hsum16(p0 + p1);
    }
    // P: write bf16 directly into per-wave A-swizzled LDS buffer, read back as one vector
#pragma unroll
    for (int sub = 0; sub < 2; ++sub)
#pragma unroll
      for (int r = 0; r < 8; ++r) {
        int off = ((r + 8 * h) + 16 * sub) * 16 + ml;
        tp[off] = (__bf16)sv[sub][r];
      }
    v16bf pA = load_sw(tp, lane);
    // P x V (K = 32 keys)
#pragma unroll
    for (int j = 0; j < 6; ++j) {
      v16bf vB = load_sw(&ldsV[cur][(size_t)j * 512], lane);
      O[j] = wmma_bf16(pA, vB, O[j]);
    }
    // next buffer must be complete, and all waves done with this one, before flipping
    if (ck + 1 < CH_NP) {
      wait_asynccnt0();
      __syncthreads();
    }
  }
  // normalize and store sim_new^T in B-swizzled layout for the ctx GEMM
#pragma unroll
  for (int r = 0; r < 8; ++r) {
    float inv = 1.f / run_l[r];
    int q = mw + r + 8 * h;
#pragma unroll
    for (int j = 0; j < 6; ++j) {
      size_t off = ((size_t)(b * 3 + (j >> 1)) * NT_NP + (q >> 4)) * 512 +
                   ((q & 15) + 16 * (j & 1)) * 16 + ml;
      snTB[off] = (__bf16)(O[j][r] * inv);
    }
  }
}

// ---------------- kernel 6: fused ctx (K=96) + final 1x1 conv ----------------
__global__ __launch_bounds__(256) void k6_out(
    const __bf16* __restrict__ v2A, const __bf16* __restrict__ snTB,
    const __bf16* __restrict__ WwA, const float* __restrict__ bw,
    float* __restrict__ out) {
  __shared__ __bf16 lctx[32 * 512];  // [kc(8)][nt(4)][512] B-swizzled ctx tiles (32 KB)
  int b = blockIdx.y, n0 = blockIdx.x * 64, tid = threadIdx.x;
  int wave = tid >> 5, lane = tid & 31, ml = lane & 15, h = lane >> 4;
  // phase A: ctx[256 x 64] = v2[256 x 96] x snT[96 x 64] -> LDS bf16 (B-swizzled)
  {
    int mt0 = wave * 2;
    v16bf aC[2][3];
#pragma unroll
    for (int mt = 0; mt < 2; ++mt)
#pragma unroll
      for (int kc = 0; kc < 3; ++kc)
        aC[mt][kc] = load_sw(v2A + ((size_t)(b * 16 + mt0 + mt) * 3 + kc) * 512, lane);
#pragma unroll
    for (int nt = 0; nt < 4; ++nt) {
      v16bf bB[3];
#pragma unroll
      for (int kc = 0; kc < 3; ++kc)
        bB[kc] = load_sw(snTB + ((size_t)(b * 3 + kc) * NT_NP + (n0 >> 4) + nt) * 512, lane);
#pragma unroll
      for (int mt = 0; mt < 2; ++mt) {
        v8f c = {};
#pragma unroll
        for (int kc = 0; kc < 3; ++kc) c = wmma_bf16(aC[mt][kc], bB[kc], c);
#pragma unroll
        for (int r = 0; r < 8; ++r) {
          int ch = wave * 32 + mt * 16 + r + 8 * h;
          lctx[((ch >> 5) * 4 + nt) * 512 + b_sw_idx(ch & 31, ml)] = (__bf16)c[r];
        }
      }
    }
  }
  __syncthreads();
  // phase B: out[512 x 64] = Ww[512 x 256] x ctx[256 x 64]
  int mt0 = wave * 4;
  v8f acc[4][4] = {};
#pragma unroll 1
  for (int kc = 0; kc < 8; ++kc) {
    v16bf aW[4];
#pragma unroll
    for (int mt = 0; mt < 4; ++mt)
      aW[mt] = load_sw(WwA + (size_t)((mt0 + mt) * 8 + kc) * 512, lane);
#pragma unroll
    for (int nt = 0; nt < 4; ++nt) {
      v16bf bB = load_sw(lctx + (size_t)(kc * 4 + nt) * 512, lane);
#pragma unroll
      for (int mt = 0; mt < 4; ++mt) acc[mt][nt] = wmma_bf16(aW[mt], bB, acc[mt][nt]);
    }
  }
#pragma unroll
  for (int mt = 0; mt < 4; ++mt)
#pragma unroll
    for (int r = 0; r < 8; ++r) {
      int o = wave * 64 + mt * 16 + r + 8 * h;
      float bo = bw[o];
#pragma unroll
      for (int nt = 0; nt < 4; ++nt) {
        int n = n0 + nt * 16 + ml;
        if (n < NSP) out[((size_t)b * C_O + o) * NSP + n] = acc[mt][nt][r] + bo;
      }
    }
}

// ---------------- host launcher ----------------
extern "C" void kernel_launch(void* const* d_in, const int* in_sizes, int n_in,
                              void* d_out, int out_size, void* d_ws, size_t ws_size,
                              hipStream_t stream) {
  (void)in_sizes; (void)n_in; (void)out_size;
  if (ws_size < OFF_END) return;
  const float* x     = (const float*)d_in[0];
  const float* Wk    = (const float*)d_in[1];
  const float* bk    = (const float*)d_in[2];
  const float* gamma = (const float*)d_in[3];
  const float* beta  = (const float*)d_in[4];
  const float* Wv    = (const float*)d_in[5];
  const float* bv    = (const float*)d_in[6];
  const float* Wv2   = (const float*)d_in[7];
  const float* bv2   = (const float*)d_in[8];
  const float* Ww    = (const float*)d_in[9];
  const float* bw    = (const float*)d_in[10];
  float* out = (float*)d_out;
  char* ws = (char*)d_ws;

  __bf16* WkA   = (__bf16*)(ws + OFF_WKA);
  __bf16* WvA   = (__bf16*)(ws + OFF_WVA);
  __bf16* Wv2A  = (__bf16*)(ws + OFF_WV2A);
  __bf16* WwA   = (__bf16*)(ws + OFF_WWA);
  float*  valf  = (float*) (ws + OFF_VALF);
  __bf16* vAsw  = (__bf16*)(ws + OFF_VA);
  __bf16* qkAsw = (__bf16*)(ws + OFF_QKA);
  __bf16* qkBsw = (__bf16*)(ws + OFF_QKB);
  __bf16* poolB = (__bf16*)(ws + OFF_POOL);
  __bf16* v2B   = (__bf16*)(ws + OFF_V2B);
  __bf16* v2A   = (__bf16*)(ws + OFF_V2A);
  __bf16* simvB = (__bf16*)(ws + OFF_SIMV);
  __bf16* snTB  = (__bf16*)(ws + OFF_SNT);

  k0_prep<<<512, 256, 0, stream>>>(Wk, Wv, Wv2, Ww, WkA, WvA, Wv2A, WwA);
  k0_zero<<<2048, 256, 0, stream>>>((uint4*)(ws + OFF_INT), (OFF_END - OFF_INT) / 16);
  k1_qkv<<<dim3(NP / 64, BATCH), 256, 0, stream>>>(x, WvA, WkA, bv, bk, gamma, beta,
                                                   valf, vAsw, qkAsw, qkBsw);
  k2_pool<<<(BATCH * C_I * KP + 255) / 256, 256, 0, stream>>>(valf, poolB);
  k3_v2<<<BATCH, 256, 0, stream>>>(Wv2A, poolB, bv2, v2B, v2A);
  k4_simv<<<dim3(NP / 64, BATCH), 128, 0, stream>>>(vAsw, v2B, simvB);
  k5_attn<<<dim3(NP / 64, BATCH), 128, 0, stream>>>(qkAsw, qkBsw, simvB, snTB);
  k6_out<<<dim3(NP / 64, BATCH), 256, 0, stream>>>(v2A, snTB, WwA, bw, out);
}
